// MambaDecoderLayer_49435073577376
// MI455X (gfx1250) — matrix operand order repaired and checked by reading, compiled, not run on hardware
//
#include <hip/hip_runtime.h>
#include <hip/hip_bf16.h>
#include <math.h>

// ---------------------------------------------------------------------------
// Model constants (match reference)
// ---------------------------------------------------------------------------
#define D_MODEL 512
#define D_INNER 1024
#define D_STATE 16
#define D_CONV 4
#define DT_RANK 32
#define D_FFN 2048
#define BATCH 8
#define SEQ_S 512
#define SEQ_T 1024
#define SEQ_C (SEQ_T + SEQ_S)   // 1536

// WMMA / TDM vector types (gfx1250, wave32)
typedef __attribute__((ext_vector_type(16))) __bf16 v16bf;
typedef __attribute__((ext_vector_type(8)))  float  v8f;
typedef __attribute__((ext_vector_type(4)))  unsigned int u32x4;
typedef __attribute__((ext_vector_type(8)))  unsigned int u32x8;

union FragAB { u32x4 u[2]; v16bf v; };

#define ACT_NONE 0
#define ACT_RELU 1
#define ACT_SOFTPLUS 2

__device__ __forceinline__ unsigned short f32_to_bf16_bits(float f) {
  union { float f; unsigned u; } v; v.f = f;
  unsigned r = v.u + 0x7FFFu + ((v.u >> 16) & 1u);  // round to nearest even
  return (unsigned short)(r >> 16);
}

__device__ __forceinline__ float silu_f(float x) {
  return x / (1.f + __expf(-x));
}

// ---------------------------------------------------------------------------
// f32 -> bf16 conversion (grid-stride)
// ---------------------------------------------------------------------------
__global__ void k_cvt_bf16(const float* __restrict__ in,
                           unsigned short* __restrict__ out, size_t n) {
  size_t i = (size_t)blockIdx.x * blockDim.x + threadIdx.x;
  size_t stride = (size_t)gridDim.x * blockDim.x;
  for (; i < n; i += stride) out[i] = f32_to_bf16_bits(in[i]);
}

// ---------------------------------------------------------------------------
// WMMA GEMM:  C[M,N] = act( A[M,K] * W[N,K]^T + bias[N] )
// A, W bf16 (K-major rows). C f32. lda/ldw/ldc in elements.
//
// Block = 4 waves (128 thr), block tile 128x64. Wave tile 32x64
// (2 m-subtiles x 4 n-subtiles = 8 accumulators, 8 WMMA / K-step).
// The 64x32 B tile is DMA'd into LDS by the Tensor Data Mover
// (tensor_load_to_lds, 2D descriptor, double buffered, TENSORcnt-tracked);
// each wave stages its own 16-row slice. A fragments load direct from global.
// Requires: M%128==0, N%64==0, K%32==0 (true for every GEMM in this layer).
// ---------------------------------------------------------------------------
__global__ __launch_bounds__(128)
void k_gemm_nt_bf16_wmma(const unsigned short* __restrict__ A, int lda,
                         const unsigned short* __restrict__ W, int ldw,
                         const float* __restrict__ bias,
                         float* __restrict__ C, int ldc,
                         int M, int N, int K, int act) {
  __shared__ __align__(16) unsigned short sB[2][64 * 32];  // 2 x 4KB B tiles

  const int wave  = threadIdx.x >> 5;
  const int lane  = threadIdx.x & 31;
  const int lhalf = lane >> 4;     // 0: lanes 0-15, 1: lanes 16-31
  const int l16   = lane & 15;

  const int tile_m = blockIdx.y * 128 + wave * 32;
  const int tile_n = blockIdx.x * 64;

  // --- TDM setup: this wave stages B rows [tile_n+wave*16, +16) ---
  const unsigned short* wslice = W + (size_t)(tile_n + wave * 16) * ldw;
  // low 32 bits of a generic shared pointer == LDS byte address (aperture map)
  const unsigned lds_base =
      (unsigned)(unsigned long long)(size_t)(void*)&sB[0][0] +
      (unsigned)(wave * 16 * 32 * 2);
  const unsigned lds_buf_stride = 64u * 32u * 2u;

  auto issue_tdm = [&](int buf, int kk) {
    const unsigned long long ga =
        (unsigned long long)(size_t)(const void*)(wslice + kk);
    u32x4 g0;
    g0[0] = 1u;                                    // count=1, user descriptor
    g0[1] = lds_base + (unsigned)buf * lds_buf_stride;   // lds_addr
    g0[2] = (unsigned)ga;                          // global_addr[31:0]
    g0[3] = (unsigned)(ga >> 32) | (2u << 30);     // global_addr[56:32], type=2
    u32x8 g1;
    g1[0] = (1u << 16);                            // wg_mask=0, data_size=2B
    g1[1] = ((unsigned)ldw & 0xFFFFu) << 16;       // tensor_dim0[15:0] @63:48
    g1[2] = ((unsigned)ldw >> 16) |
            (((unsigned)N & 0xFFFFu) << 16);       // dim0 hi | tensor_dim1 lo
    g1[3] = ((unsigned)N >> 16) | (32u << 16);     // dim1 hi | tile_dim0=32
    g1[4] = 16u;                                   // tile_dim1=16, tile_dim2=0
    g1[5] = (unsigned)ldw;                         // tensor_dim0_stride[31:0]
    g1[6] = 0u;                                    // stride hi | dim1_stride lo
    g1[7] = 0u;
    asm volatile("tensor_load_to_lds %0, %1" :: "s"(g0), "s"(g1) : "memory");
  };

  const v8f vzero = {0.f, 0.f, 0.f, 0.f, 0.f, 0.f, 0.f, 0.f};
  v8f acc[2][4];
#pragma unroll
  for (int ms = 0; ms < 2; ms++)
#pragma unroll
    for (int nt = 0; nt < 4; nt++) acc[ms][nt] = vzero;

  // A fragment bases: lane holds K runs [lhalf*8,+8) and [16+lhalf*8,+8)
  const unsigned short* Ap0 = A + (size_t)(tile_m + l16) * lda + lhalf * 8;
  const unsigned short* Ap1 = Ap0 + (size_t)16 * lda;

  issue_tdm(0, 0);   // prologue: DMA first B tile

  int buf = 0;
  for (int kk = 0; kk < K; kk += 32) {
    if (kk + 32 < K) {
      issue_tdm(buf ^ 1, kk + 32);                 // overlap next DMA
      __builtin_amdgcn_s_wait_tensorcnt(1);        // current tile landed
    } else {
      __builtin_amdgcn_s_wait_tensorcnt(0);
    }
    __syncthreads();                               // publish all 4 slices

    FragAB fa0, fa1;
    fa0.u[0] = *(const u32x4*)(Ap0 + kk);
    fa0.u[1] = *(const u32x4*)(Ap0 + kk + 16);
    fa1.u[0] = *(const u32x4*)(Ap1 + kk);
    fa1.u[1] = *(const u32x4*)(Ap1 + kk + 16);

    const unsigned short* bbase = &sB[buf][0];
#pragma unroll
    for (int nt = 0; nt < 4; nt++) {
      // B frag: col n = nt*16+l16, K runs [lhalf*8,+8) and [16+lhalf*8,+8) within the LDS tile
      const unsigned short* bp = bbase + (nt * 16 + l16) * 32 + lhalf * 8;
      FragAB fb;
      fb.u[0] = *(const u32x4*)(bp);
      fb.u[1] = *(const u32x4*)(bp + 16);
      acc[0][nt] = __builtin_amdgcn_wmma_f32_16x16x32_bf16(
          false, fa0.v, false, fb.v, (short)0, acc[0][nt], false, false);
      acc[1][nt] = __builtin_amdgcn_wmma_f32_16x16x32_bf16(
          false, fa1.v, false, fb.v, (short)0, acc[1][nt], false, false);
    }
    __syncthreads();                               // tile consumed, reusable
    buf ^= 1;
  }

  // Epilogue. C/D layout: VGPR r -> row = base + r + 8*lhalf, col = ntile+l16
#pragma unroll
  for (int ms = 0; ms < 2; ms++) {
#pragma unroll
    for (int nt = 0; nt < 4; nt++) {
      const int ncol = tile_n + nt * 16 + l16;
      const float bv = bias ? bias[ncol] : 0.f;
#pragma unroll
      for (int r = 0; r < 8; r++) {
        const int row = tile_m + ms * 16 + r + 8 * lhalf;
        float v = acc[ms][nt][r] + bv;
        if (act == ACT_RELU) {
          v = fmaxf(v, 0.f);
        } else if (act == ACT_SOFTPLUS) {
          v = (v > 20.f) ? v : log1pf(__expf(v));
        }
        C[(size_t)row * ldc + ncol] = v;
      }
    }
  }
}

// ---------------------------------------------------------------------------
// Depthwise causal conv (width 4) + bias + SiLU over xz[:, :D_INNER]
// ---------------------------------------------------------------------------
__global__ void k_conv_silu(const float* __restrict__ xz,
                            const float* __restrict__ cw,   // [D_INNER,4]
                            const float* __restrict__ cb,   // [D_INNER]
                            float* __restrict__ xc,
                            unsigned short* __restrict__ xc_bf,
                            int L, size_t total) {
  size_t i = (size_t)blockIdx.x * blockDim.x + threadIdx.x;
  size_t stride = (size_t)gridDim.x * blockDim.x;
  for (; i < total; i += stride) {
    const int d = (int)(i % D_INNER);
    const size_t row = i / D_INNER;
    const int l = (int)(row % L);
    float acc = cb[d];
#pragma unroll
    for (int j = 0; j < D_CONV; j++) {
      const int ls = l - (D_CONV - 1) + j;
      if (ls >= 0)
        acc += cw[d * D_CONV + j] * xz[(row - (D_CONV - 1) + j) * (2 * D_INNER) + d];
    }
    const float s = silu_f(acc);
    xc[row * D_INNER + d] = s;
    xc_bf[row * D_INNER + d] = f32_to_bf16_bits(s);
  }
}

// ---------------------------------------------------------------------------
// Selective scan. One thread per (b, d) channel, 16-state in registers.
// ---------------------------------------------------------------------------
__global__ __launch_bounds__(256)
void k_scan(const float* __restrict__ delta,  // [rows, D_INNER]
            const float* __restrict__ u,      // xc, [rows, D_INNER]
            const float* __restrict__ xdbl,   // [rows, 64]
            const float* __restrict__ A_log,  // [D_INNER, 16]
            float* __restrict__ y,            // [rows, D_INNER]
            int L) {
  __shared__ float sB[D_STATE];
  __shared__ float sC[D_STATE];
  const int b = blockIdx.x >> 2;
  const int d = ((blockIdx.x & 3) << 8) + threadIdx.x;

  float Aneg[D_STATE], h[D_STATE];
#pragma unroll
  for (int s = 0; s < D_STATE; s++) {
    Aneg[s] = -__expf(A_log[d * D_STATE + s]);
    h[s] = 0.f;
  }

  for (int l = 0; l < L; l++) {
    const size_t row = (size_t)b * L + l;
    if (threadIdx.x < 32) {
      const float v = xdbl[row * 64 + DT_RANK + threadIdx.x];
      if (threadIdx.x < D_STATE) sB[threadIdx.x] = v;
      else                       sC[threadIdx.x - D_STATE] = v;
    }
    __syncthreads();
    const float dt = delta[row * D_INNER + d];
    const float du = dt * u[row * D_INNER + d];
    float acc = 0.f;
#pragma unroll
    for (int s = 0; s < D_STATE; s++) {
      h[s] = __expf(dt * Aneg[s]) * h[s] + du * sB[s];
      acc += h[s] * sC[s];
    }
    y[row * D_INNER + d] = acc;
    __syncthreads();
  }
}

// ---------------------------------------------------------------------------
// Gate: yg = (scan_y + xc * D_vec[d]) * silu(z), z = xz[:, D_INNER + d]
// ---------------------------------------------------------------------------
__global__ void k_gate(const float* __restrict__ ys,
                       const float* __restrict__ xc,
                       const float* __restrict__ xz,
                       const float* __restrict__ Dv,
                       unsigned short* __restrict__ yg_bf, size_t total) {
  size_t i = (size_t)blockIdx.x * blockDim.x + threadIdx.x;
  size_t stride = (size_t)gridDim.x * blockDim.x;
  for (; i < total; i += stride) {
    const int d = (int)(i % D_INNER);
    const size_t row = i / D_INNER;
    const float z = xz[row * (2 * D_INNER) + D_INNER + d];
    const float v = (ys[i] + xc[i] * Dv[d]) * silu_f(z);
    yg_bf[i] = f32_to_bf16_bits(v);
  }
}

// ---------------------------------------------------------------------------
// cat = [memory, t] along sequence, stored bf16
// ---------------------------------------------------------------------------
__global__ void k_cat_bf16(const float* __restrict__ mem,   // [B, T, 512]
                           const float* __restrict__ t,     // [B, S, 512]
                           unsigned short* __restrict__ cat, size_t total) {
  size_t i = (size_t)blockIdx.x * blockDim.x + threadIdx.x;
  size_t stride = (size_t)gridDim.x * blockDim.x;
  for (; i < total; i += stride) {
    const int col = (int)(i % D_MODEL);
    const size_t row = i / D_MODEL;
    const int b = (int)(row / SEQ_C);
    const int r = (int)(row % SEQ_C);
    float v;
    if (r < SEQ_T) v = mem[((size_t)b * SEQ_T + r) * D_MODEL + col];
    else           v = t[((size_t)b * SEQ_S + (r - SEQ_T)) * D_MODEL + col];
    cat[i] = f32_to_bf16_bits(v);
  }
}

// ---------------------------------------------------------------------------
// out = LayerNorm(x + add), row length 512; 1 block = 1 row
// add row index = b*addBatchStride + addRowOff + l
// ---------------------------------------------------------------------------
__global__ __launch_bounds__(256)
void k_add_ln(const float* __restrict__ x,
              const float* __restrict__ add,
              const float* __restrict__ g, const float* __restrict__ bta,
              float* __restrict__ out, unsigned short* __restrict__ out_bf,
              int S, int addBatchStride, int addRowOff) {
  __shared__ float red[256];
  const int row = blockIdx.x;
  const int b = row / S, l = row % S;
  const size_t arow = (size_t)b * addBatchStride + addRowOff + l;
  const int c0 = threadIdx.x * 2;

  float v0 = x[(size_t)row * D_MODEL + c0]     + add[arow * D_MODEL + c0];
  float v1 = x[(size_t)row * D_MODEL + c0 + 1] + add[arow * D_MODEL + c0 + 1];

  red[threadIdx.x] = v0 + v1;
  __syncthreads();
  for (int o = 128; o > 0; o >>= 1) {
    if (threadIdx.x < o) red[threadIdx.x] += red[threadIdx.x + o];
    __syncthreads();
  }
  const float mu = red[0] * (1.f / D_MODEL);
  __syncthreads();

  const float d0 = v0 - mu, d1 = v1 - mu;
  red[threadIdx.x] = d0 * d0 + d1 * d1;
  __syncthreads();
  for (int o = 128; o > 0; o >>= 1) {
    if (threadIdx.x < o) red[threadIdx.x] += red[threadIdx.x + o];
    __syncthreads();
  }
  const float rstd = rsqrtf(red[0] * (1.f / D_MODEL) + 1e-6f);

  const float o0 = d0 * rstd * g[c0]     + bta[c0];
  const float o1 = d1 * rstd * g[c0 + 1] + bta[c0 + 1];
  out[(size_t)row * D_MODEL + c0]     = o0;
  out[(size_t)row * D_MODEL + c0 + 1] = o1;
  if (out_bf) {
    out_bf[(size_t)row * D_MODEL + c0]     = f32_to_bf16_bits(o0);
    out_bf[(size_t)row * D_MODEL + c0 + 1] = f32_to_bf16_bits(o1);
  }
}

// ---------------------------------------------------------------------------
// Host side
// ---------------------------------------------------------------------------
static inline void* carve(char*& p, size_t bytes) {
  void* r = (void*)p;
  p += (bytes + 255) & ~(size_t)255;
  return r;
}

static inline void launch_cvt(const float* in, unsigned short* out, size_t n,
                              hipStream_t s) {
  int blocks = (int)((n + 255) / 256);
  if (blocks > 4096) blocks = 4096;
  hipLaunchKernelGGL(k_cvt_bf16, dim3(blocks), dim3(256), 0, s, in, out, n);
}

static inline void launch_gemm(const unsigned short* A, int lda,
                               const unsigned short* W, int ldw,
                               const float* bias, float* C, int ldc,
                               int M, int N, int K, int act, hipStream_t s) {
  dim3 grid(N / 64, M / 128);
  hipLaunchKernelGGL(k_gemm_nt_bf16_wmma, grid, dim3(128), 0, s,
                     A, lda, W, ldw, bias, C, ldc, M, N, K, act);
}

struct MambaScratch {
  float* xz;             // [rows, 2048]
  float* xc;             // [rows, 1024]
  unsigned short* xc_bf; // [rows, 1024]
  float* xdbl;           // [rows, 64]
  unsigned short* xdbl_bf;
  float* delta;          // [rows, 1024]
  float* ys;             // [rows, 1024]
  unsigned short* yg_bf; // [rows, 1024]
};

static void run_mamba(hipStream_t s, int L, const unsigned short* in_bf,
                      const unsigned short* w_in_bf,
                      const float* conv_w, const float* conv_b,
                      const unsigned short* w_xp_bf,
                      const unsigned short* w_dt_bf, const float* dt_b,
                      const float* A_log, const float* D_vec,
                      const unsigned short* w_out_bf,
                      const MambaScratch& sc, float* mo) {
  const int rows = BATCH * L;
  const size_t rDI = (size_t)rows * D_INNER;

  // in_proj: xz = x @ in_w^T        [rows,2048]
  launch_gemm(in_bf, D_MODEL, w_in_bf, D_MODEL, nullptr,
              sc.xz, 2 * D_INNER, rows, 2 * D_INNER, D_MODEL, ACT_NONE, s);
  // depthwise causal conv + SiLU
  hipLaunchKernelGGL(k_conv_silu, dim3(2048), dim3(256), 0, s,
                     sc.xz, conv_w, conv_b, sc.xc, sc.xc_bf, L, rDI);
  // x_proj: x_dbl = xc @ xproj_w^T  [rows,64]
  launch_gemm(sc.xc_bf, D_INNER, w_xp_bf, D_INNER, nullptr,
              sc.xdbl, 64, rows, 64, D_INNER, ACT_NONE, s);
  launch_cvt(sc.xdbl, sc.xdbl_bf, (size_t)rows * 64, s);
  // dt_proj + softplus: delta = softplus(dt @ dt_w^T + dt_b)  (K=32, lda=64)
  launch_gemm(sc.xdbl_bf, 64, w_dt_bf, DT_RANK, dt_b,
              sc.delta, D_INNER, rows, D_INNER, DT_RANK, ACT_SOFTPLUS, s);
  // selective scan
  hipLaunchKernelGGL(k_scan, dim3(BATCH * 4), dim3(256), 0, s,
                     sc.delta, sc.xc, sc.xdbl, A_log, sc.ys, L);
  // gate
  hipLaunchKernelGGL(k_gate, dim3(2048), dim3(256), 0, s,
                     sc.ys, sc.xc, sc.xz, D_vec, sc.yg_bf, rDI);
  // out_proj: mo = yg @ out_w^T     [rows,512]
  launch_gemm(sc.yg_bf, D_INNER, w_out_bf, D_INNER, nullptr,
              mo, D_MODEL, rows, D_MODEL, D_INNER, ACT_NONE, s);
}

extern "C" void kernel_launch(void* const* d_in, const int* in_sizes, int n_in,
                              void* d_out, int out_size, void* d_ws, size_t ws_size,
                              hipStream_t stream) {
  (void)in_sizes; (void)n_in; (void)out_size; (void)ws_size;

  const float* tgt    = (const float*)d_in[0];
  const float* memory = (const float*)d_in[1];
  const float* s_in_w   = (const float*)d_in[2];
  const float* s_conv_w = (const float*)d_in[3];
  const float* s_conv_b = (const float*)d_in[4];
  const float* s_xp_w   = (const float*)d_in[5];
  const float* s_dt_w   = (const float*)d_in[6];
  const float* s_dt_b   = (const float*)d_in[7];
  const float* s_A_log  = (const float*)d_in[8];
  const float* s_D_vec  = (const float*)d_in[9];
  const float* s_out_w  = (const float*)d_in[10];
  const float* c_in_w   = (const float*)d_in[11];
  const float* c_conv_w = (const float*)d_in[12];
  const float* c_conv_b = (const float*)d_in[13];
  const float* c_xp_w   = (const float*)d_in[14];
  const float* c_dt_w   = (const float*)d_in[15];
  const float* c_dt_b   = (const float*)d_in[16];
  const float* c_A_log  = (const float*)d_in[17];
  const float* c_D_vec  = (const float*)d_in[18];
  const float* c_out_w  = (const float*)d_in[19];
  const float* ln1_g = (const float*)d_in[20];
  const float* ln1_b = (const float*)d_in[21];
  const float* ln2_g = (const float*)d_in[22];
  const float* ln2_b = (const float*)d_in[23];
  const float* ln3_g = (const float*)d_in[24];
  const float* ln3_b = (const float*)d_in[25];
  const float* ffn_w1 = (const float*)d_in[26];
  const float* ffn_b1 = (const float*)d_in[27];
  const float* ffn_w2 = (const float*)d_in[28];
  const float* ffn_b2 = (const float*)d_in[29];

  const int Msel = BATCH * SEQ_S;   // 4096
  const int Mcat = BATCH * SEQ_C;   // 12288

  char* p = (char*)d_ws;

  // bf16 weight copies
  unsigned short* wbf_s_in  = (unsigned short*)carve(p, (size_t)2*D_INNER*D_MODEL*2);
  unsigned short* wbf_s_xp  = (unsigned short*)carve(p, (size_t)64*D_INNER*2);
  unsigned short* wbf_s_dt  = (unsigned short*)carve(p, (size_t)D_INNER*DT_RANK*2);
  unsigned short* wbf_s_out = (unsigned short*)carve(p, (size_t)D_MODEL*D_INNER*2);
  unsigned short* wbf_c_in  = (unsigned short*)carve(p, (size_t)2*D_INNER*D_MODEL*2);
  unsigned short* wbf_c_xp  = (unsigned short*)carve(p, (size_t)64*D_INNER*2);
  unsigned short* wbf_c_dt  = (unsigned short*)carve(p, (size_t)D_INNER*DT_RANK*2);
  unsigned short* wbf_c_out = (unsigned short*)carve(p, (size_t)D_MODEL*D_INNER*2);
  unsigned short* wbf_f1    = (unsigned short*)carve(p, (size_t)D_FFN*D_MODEL*2);
  unsigned short* wbf_f2    = (unsigned short*)carve(p, (size_t)D_MODEL*D_FFN*2);

  // activations (sized for the larger cross pass, reused by both)
  unsigned short* tgt_bf = (unsigned short*)carve(p, (size_t)Msel*D_MODEL*2);
  MambaScratch sc;
  sc.xz      = (float*)carve(p, (size_t)Mcat*2*D_INNER*4);
  sc.xc      = (float*)carve(p, (size_t)Mcat*D_INNER*4);
  sc.xc_bf   = (unsigned short*)carve(p, (size_t)Mcat*D_INNER*2);
  sc.xdbl    = (float*)carve(p, (size_t)Mcat*64*4);
  sc.xdbl_bf = (unsigned short*)carve(p, (size_t)Mcat*64*2);
  sc.delta   = (float*)carve(p, (size_t)Mcat*D_INNER*4);
  sc.ys      = (float*)carve(p, (size_t)Mcat*D_INNER*4);
  sc.yg_bf   = (unsigned short*)carve(p, (size_t)Mcat*D_INNER*2);
  float* mo  = (float*)carve(p, (size_t)Mcat*D_MODEL*4);
  float* t1  = (float*)carve(p, (size_t)Msel*D_MODEL*4);
  unsigned short* t1_bf = (unsigned short*)carve(p, (size_t)Msel*D_MODEL*2);
  unsigned short* cat_bf = (unsigned short*)carve(p, (size_t)Mcat*D_MODEL*2);
  float* t2  = (float*)carve(p, (size_t)Msel*D_MODEL*4);
  unsigned short* t2_bf = (unsigned short*)carve(p, (size_t)Msel*D_MODEL*2);
  // FFN reuses big mamba buffers (mamba phases are finished by then)
  float* f = sc.xz;                                  // [4096,2048] f32
  unsigned short* f_bf = sc.yg_bf;                   // [4096,2048] bf16
  float* ffn_out = sc.xc;                            // [4096,512]  f32

  // ---- convert weights + input to bf16 ----
  launch_cvt(s_in_w,  wbf_s_in,  (size_t)2*D_INNER*D_MODEL, stream);
  launch_cvt(s_xp_w,  wbf_s_xp,  (size_t)64*D_INNER,        stream);
  launch_cvt(s_dt_w,  wbf_s_dt,  (size_t)D_INNER*DT_RANK,   stream);
  launch_cvt(s_out_w, wbf_s_out, (size_t)D_MODEL*D_INNER,   stream);
  launch_cvt(c_in_w,  wbf_c_in,  (size_t)2*D_INNER*D_MODEL, stream);
  launch_cvt(c_xp_w,  wbf_c_xp,  (size_t)64*D_INNER,        stream);
  launch_cvt(c_dt_w,  wbf_c_dt,  (size_t)D_INNER*DT_RANK,   stream);
  launch_cvt(c_out_w, wbf_c_out, (size_t)D_MODEL*D_INNER,   stream);
  launch_cvt(ffn_w1,  wbf_f1,    (size_t)D_FFN*D_MODEL,     stream);
  launch_cvt(ffn_w2,  wbf_f2,    (size_t)D_MODEL*D_FFN,     stream);
  launch_cvt(tgt,     tgt_bf,    (size_t)Msel*D_MODEL,      stream);

  // ---- self mamba ----
  run_mamba(stream, SEQ_S, tgt_bf, wbf_s_in, s_conv_w, s_conv_b,
            wbf_s_xp, wbf_s_dt, s_dt_b, s_A_log, s_D_vec, wbf_s_out, sc, mo);
  // t1 = LN1(tgt + mo)
  hipLaunchKernelGGL(k_add_ln, dim3(Msel), dim3(256), 0, stream,
                     tgt, mo, ln1_g, ln1_b, t1, t1_bf, SEQ_S, SEQ_S, 0);

  // ---- cross mamba on cat([memory, t1]) ----
  hipLaunchKernelGGL(k_cat_bf16, dim3(2048), dim3(256), 0, stream,
                     memory, t1, cat_bf, (size_t)Mcat * D_MODEL);
  run_mamba(stream, SEQ_C, cat_bf, wbf_c_in, c_conv_w, c_conv_b,
            wbf_c_xp, wbf_c_dt, c_dt_b, c_A_log, c_D_vec, wbf_c_out, sc, mo);
  // t2 = LN2(t1 + mo[:, -S:])   (cross rows: b*1536 + 1024 + l)
  hipLaunchKernelGGL(k_add_ln, dim3(Msel), dim3(256), 0, stream,
                     t1, mo, ln2_g, ln2_b, t2, t2_bf, SEQ_S, SEQ_C, SEQ_T);

  // ---- FFN ----
  launch_gemm(t2_bf, D_MODEL, wbf_f1, D_MODEL, ffn_b1,
              f, D_FFN, Msel, D_FFN, D_MODEL, ACT_RELU, stream);
  launch_cvt(f, f_bf, (size_t)Msel * D_FFN, stream);
  launch_gemm(f_bf, D_FFN, wbf_f2, D_FFN, ffn_b2,
              ffn_out, D_MODEL, Msel, D_MODEL, D_FFN, ACT_NONE, stream);
  // out = LN3(t2 + ffn_out)
  hipLaunchKernelGGL(k_add_ln, dim3(Msel), dim3(256), 0, stream,
                     t2, ffn_out, ln3_g, ln3_b, (float*)d_out, (unsigned short*)nullptr,
                     SEQ_S, SEQ_S, 0);
}
